// AttentiveLSTMCell_16277926052471
// MI455X (gfx1250) — matrix operand order repaired and checked by reading, not compiled, not run on hardware
//
#include <hip/hip_runtime.h>
#include <math.h>

typedef unsigned short u16;
typedef unsigned int u32;
typedef __attribute__((ext_vector_type(16))) __bf16 v16bf;
typedef __attribute__((ext_vector_type(8)))  float  v8f;
typedef __attribute__((ext_vector_type(4)))  unsigned int v4u;
typedef __attribute__((ext_vector_type(8)))  int v8i;
typedef __attribute__((ext_vector_type(4)))  int v4i;

union FragU {
    v16bf bf;
    u16   u[16];
    uint4 q[2];
};

// Native f32 -> bf16 (v_cvt_bf16_f32 / v_cvt_pk_bf16_f32, RNE on gfx1250)
__device__ __forceinline__ u16 f2bf(float f) {
    return __builtin_bit_cast(u16, (__bf16)f);
}
// pack two converts into one dword (compiler can fuse into v_cvt_pk_bf16_f32)
__device__ __forceinline__ u32 pk2bf(float x, float y) {
    return (u32)f2bf(x) | ((u32)f2bf(y) << 16);
}
__device__ __forceinline__ float bflo(u32 v) { return __uint_as_float(v << 16); }
__device__ __forceinline__ float bfhi(u32 v) { return __uint_as_float(v & 0xFFFF0000u); }
__device__ __forceinline__ float hsig(float x) { return fminf(fmaxf(0.2f * x + 0.5f, 0.f), 1.f); }

// Hardware tanh (V_TANH_F32 transcendental; co-executes with WMMA)
__device__ __forceinline__ float fast_tanh(float x) {
#if __has_builtin(__builtin_amdgcn_tanhf)
    return __builtin_amdgcn_tanhf(x);
#elif __has_builtin(__builtin_amdgcn_tanh_f32)
    return __builtin_amdgcn_tanh_f32(x);
#else
    return tanhf(x);
#endif
}

__device__ __forceinline__ v8f wmma_bf16(v16bf a, v16bf b, v8f c) {
    return __builtin_amdgcn_wmma_f32_16x16x32_bf16(false, a, false, b, (short)0, c, false, false);
}

#define MNEG (-3.0e38f)

// ---------------------------------------------------------------------------
// TDM staging (one 2KB contiguous slice per wave), with fallback copy.
// 6-arg builtin: (uint32x4 g0, int32x8 g1, int32x4 g2, int32x4 g3, int32x8, i32 cpol)
// ---------------------------------------------------------------------------
#if __has_builtin(__builtin_amdgcn_tensor_load_to_lds)
#define HAVE_TDM 1
__device__ __forceinline__ void stage_issue(u16* lds_dst, const u16* gsrc) {
    unsigned long long ga = (unsigned long long)(size_t)gsrc;
    u32 lds_addr = (u32)(size_t)lds_dst;          // low 32 bits = LDS byte offset
    v4u g0;
    g0.x = 1u;                                     // count=1, user descriptor
    g0.y = lds_addr;                               // lds_addr
    g0.z = (u32)ga;                                // global_addr[31:0]
    g0.w = (u32)((ga >> 32) & 0x1FFFFFFu) | (2u << 30); // addr[56:32] | type=2
    v8i g1;
    g1[0] = 0x00020000;    // wg_mask=0, data_size=4B
    g1[1] = 0;             // atomic barrier addr=0, tensor_dim0[15:0]=0
    g1[2] = 0x00010100;    // tensor_dim0=1<<24, tensor_dim1=1
    g1[3] = 0x02000000;    // tile_dim0=512 dwords
    g1[4] = 1;             // tile_dim1=1, tile_dim2=0
    g1[5] = 512;           // tensor_dim0_stride
    g1[6] = 0;
    g1[7] = 0;
    v4i z4 = (v4i)0;
    v8i z8 = (v8i)0;
    __builtin_amdgcn_tensor_load_to_lds(g0, g1, z4, z4, z8, 0);
}
__device__ __forceinline__ void stage_wait() { __builtin_amdgcn_s_wait_tensorcnt(0); }
#else
#define HAVE_TDM 0
__device__ __forceinline__ void stage_issue(u16* lds_dst, const u16* gsrc) {
    const uint4* s = (const uint4*)gsrc;
    uint4* d = (uint4*)lds_dst;
    int l = threadIdx.x & 31;
    d[l] = s[l]; d[l + 32] = s[l + 32]; d[l + 64] = s[l + 64]; d[l + 96] = s[l + 96];
}
__device__ __forceinline__ void stage_wait() {}
#endif

// ---------------------------------------------------------------------------
// Pack f32 weights [K][N] (optionally concat of two sources split at Ksplit)
// into bf16 WMMA B-fragment layout.
// frag = ((p*G + g)*4 + kc)*4 + nb ; per frag: 32 lanes x 16 elems.
//   n = p*64 + nb*16 + (lane&15)
//   k = g*128 + kc*32 + 8*(lane>>4) + (e&7) + 16*(e>>3)   (B-matrix layout)
// ---------------------------------------------------------------------------
__global__ void pack_kernel(const float* __restrict__ A, const float* __restrict__ B,
                            int Ksplit, int N, int G, u16* __restrict__ dst, int total) {
    int idx = blockIdx.x * 256 + threadIdx.x;
    if (idx >= total) return;
    int r = idx & 511, frag = idx >> 9;
    int lane = r >> 4, e = r & 15;
    int nb = frag & 3;
    int t1 = frag >> 2;
    int kc = t1 & 3;
    int t2 = t1 >> 2;
    int g = t2 % G, p = t2 / G;
    int n = p * 64 + nb * 16 + (lane & 15);
    int k = g * 128 + kc * 32 + 8 * (lane >> 4) + (e & 7) + 16 * (e >> 3);
    float v = (k < Ksplit) ? A[(size_t)k * N + n] : B[(size_t)(k - Ksplit) * N + n];
    dst[idx] = f2bf(v);
}

// A-fragment element mapping (16-bit A 16x32):
//   e=0..7  -> k = 8*(lane>>4) + e
//   e=8..15 -> k = 16 + 8*(lane>>4) + (e-8)
__device__ __forceinline__ void load_afrag_f32(FragU& fa, const float* row, int hi) {
    const float* p0 = row + 8 * hi;
#pragma unroll
    for (int e = 0; e < 4; ++e) {
        u32 lo = pk2bf(p0[2 * e], p0[2 * e + 1]);
        u32 hi2 = pk2bf(p0[2 * e + 16], p0[2 * e + 17]);
        ((u32*)&fa)[e] = lo;
        ((u32*)&fa)[e + 4] = hi2;
    }
}

// ---------------------------------------------------------------------------
// Wx = h @ kernel_w   (64x512x512), one wave per 16x16 tile
// ---------------------------------------------------------------------------
__global__ __launch_bounds__(32) void wx_kernel(const float* __restrict__ h,
                                                const u16* __restrict__ pk_kw,
                                                float* __restrict__ wx) {
    int l = threadIdx.x, l15 = l & 15, hi = l >> 4;
    int bx = blockIdx.x, by = blockIdx.y;
    int p = bx >> 2, nb = bx & 3;
    v8f c = {};
#pragma unroll
    for (int g = 0; g < 4; ++g)
#pragma unroll
        for (int kc = 0; kc < 4; ++kc) {
            int kb = g * 128 + kc * 32;
            FragU fa; load_afrag_f32(fa, h + (size_t)(by * 16 + l15) * 512 + kb, hi);
            int fi = ((p * 4 + g) * 4 + kc) * 4 + nb;
            const uint4* bp = (const uint4*)((const char*)pk_kw + (size_t)fi * 1024 + l * 32);
            FragU fb; fb.q[0] = bp[0]; fb.q[1] = bp[1];
            c = wmma_bf16(fa.bf, fb.bf, c);
        }
    int col = bx * 16 + l15;
#pragma unroll
    for (int j = 0; j < 8; ++j) wx[(size_t)(by * 16 + j + 8 * hi) * 512 + col] = c[j];
}

// ---------------------------------------------------------------------------
// Fused flash-attention energy + online softmax + context partial.
// Grid: (chunk=4, batch=64). 256 threads = 8 waves. Each WG: 512 t-rows,
// processed as 4 tiles of 128 rows, ann tile double-buffered bf16 in LDS.
// ---------------------------------------------------------------------------
__global__ __launch_bounds__(256) void attn_kernel(const float* __restrict__ ann,
                                                   const float* __restrict__ wx,
                                                   const float* __restrict__ bias,
                                                   const float* __restrict__ kv,
                                                   const u16* __restrict__ pk_ku,
                                                   float* __restrict__ partials) {
    __shared__ u16  ann_s[2][128 * 520];
    __shared__ u16  bstage[2][8192];
    __shared__ float wxb_s[512];
    __shared__ float v_s[512];
    __shared__ float et_s[128];
    __shared__ float p_s[128];
    __shared__ float red_s[8];

    const int tid = threadIdx.x;
    const int l = tid & 31, wv = tid >> 5;
    const int l15 = l & 15, hi = l >> 4;
    const int b = blockIdx.y, chunk = blockIdx.x;
    const float* annb = ann + (size_t)(b * 2048 + chunk * 512) * 512;

    for (int u = tid; u < 512; u += 256) {
        wxb_s[u] = wx[b * 512 + u] + bias[2048 + u];
        v_s[u] = kv[u];
    }
    // preload tile 0 (128 rows x 512 f32 -> bf16)
    {
        const float4* g4 = (const float4*)annb;
#pragma unroll 4
        for (int q = 0; q < 64; ++q) {
            int idx = q * 256 + tid;
            int row = idx >> 7, col = (idx & 127) * 4;
            float4 f = g4[idx];
            uint2 s;
            s.x = pk2bf(f.x, f.y);
            s.y = pk2bf(f.z, f.w);
            *(uint2*)&ann_s[0][row * 520 + col] = s;
        }
    }
    // B stage 0 -> buffer 0 (this wave's 2KB slice)
    stage_issue(&bstage[0][wv * 1024], pk_ku + wv * 1024);

    float m_run = MNEG, l_run = 0.f, c0 = 0.f, c1 = 0.f;
    int s_lin = 0;

    for (int tile = 0; tile < 4; ++tile) {
        const int cur = tile & 1, nxt = cur ^ 1;
        const bool do_pf = (tile < 3);
        float et_lp[8];
#pragma unroll
        for (int j = 0; j < 8; ++j) et_lp[j] = 0.f;

        for (int p = 0; p < 8; ++p) {
            // prefetch slice p of next tile into registers (overlaps WMMA)
            float4 pf[8];
            if (do_pf) {
                const float4* g4 = (const float4*)(annb + (size_t)(tile + 1) * 128 * 512);
#pragma unroll
                for (int q = 0; q < 8; ++q) pf[q] = g4[p * 2048 + q * 256 + tid];
            }
            v8f cc[4];
#pragma unroll
            for (int nb = 0; nb < 4; ++nb) cc[nb] = (v8f){};

            for (int g = 0; g < 4; ++g) {
                stage_wait();
                __syncthreads();
                int ns = s_lin + 1;
                stage_issue(&bstage[ns & 1][wv * 1024],
                            pk_ku + (size_t)(ns & 31) * 8192 + wv * 1024);
                const u16* bs = &bstage[s_lin & 1][0];
#pragma unroll
                for (int kc = 0; kc < 4; ++kc) {
                    int k0 = g * 128 + kc * 32 + 8 * hi;
                    int ro = (wv * 16 + l15) * 520 + k0;
                    FragU fa;
                    fa.q[0] = *(const uint4*)&ann_s[cur][ro];
                    fa.q[1] = *(const uint4*)&ann_s[cur][ro + 16];
#pragma unroll
                    for (int nb = 0; nb < 4; ++nb) {
                        const uint4* bp = (const uint4*)&bs[(kc * 4 + nb) * 512 + l * 16];
                        FragU fb; fb.q[0] = bp[0]; fb.q[1] = bp[1];
                        cc[nb] = wmma_bf16(fa.bf, fb.bf, cc[nb]);
                    }
                }
                ++s_lin;
            }
            // panel epilogue: et partial += tanh(uh + wx + b) * v
#pragma unroll
            for (int nb = 0; nb < 4; ++nb) {
                int col = p * 64 + nb * 16 + l15;
                float wxv = wxb_s[col], vv = v_s[col];
#pragma unroll
                for (int j = 0; j < 8; ++j)
                    et_lp[j] += fast_tanh(cc[nb][j] + wxv) * vv;
            }
            // commit prefetched slice to next buffer
            if (do_pf) {
#pragma unroll
                for (int q = 0; q < 8; ++q) {
                    int idx = p * 2048 + q * 256 + tid;
                    int row = idx >> 7, col = (idx & 127) * 4;
                    uint2 s;
                    s.x = pk2bf(pf[q].x, pf[q].y);
                    s.y = pk2bf(pf[q].z, pf[q].w);
                    *(uint2*)&ann_s[nxt][row * 520 + col] = s;
                }
            }
        }
        // reduce et over columns (within 16-lane halves) and store
#pragma unroll
        for (int j = 0; j < 8; ++j) {
            float v = et_lp[j];
            v += __shfl_xor(v, 1, 32);
            v += __shfl_xor(v, 2, 32);
            v += __shfl_xor(v, 4, 32);
            v += __shfl_xor(v, 8, 32);
            if (l15 == 0) et_s[wv * 16 + j + 8 * hi] = v;
        }
        __syncthreads();

        // online softmax update for this tile
        float mv = (tid < 128) ? et_s[tid] : MNEG;
        for (int msk = 16; msk > 0; msk >>= 1) mv = fmaxf(mv, __shfl_xor(mv, msk, 32));
        if (l == 0) red_s[wv] = mv;
        __syncthreads();
        float mt = red_s[0];
#pragma unroll
        for (int w = 1; w < 8; ++w) mt = fmaxf(mt, red_s[w]);
        float m_new = fmaxf(m_run, mt);
        float pp = 0.f;
        if (tid < 128) { pp = __expf(et_s[tid] - m_new); p_s[tid] = pp; }
        __syncthreads();
        float sv = pp;
        for (int msk = 16; msk > 0; msk >>= 1) sv += __shfl_xor(sv, msk, 32);
        if (l == 0) red_s[wv] = sv;
        __syncthreads();
        float sp = red_s[0];
#pragma unroll
        for (int w = 1; w < 8; ++w) sp += red_s[w];
        float sc = __expf(m_run - m_new);
        l_run = l_run * sc + sp;
        m_run = m_new;
        c0 *= sc; c1 *= sc;
        // context partial: each thread owns columns (2t, 2t+1)
        for (int i = 0; i < 128; ++i) {
            float pi = p_s[i];
            u32 vb = *(const u32*)&ann_s[cur][i * 520 + 2 * tid];
            c0 += pi * bflo(vb);
            c1 += pi * bfhi(vb);
        }
        __syncthreads();
    }
    float* pb = partials + (size_t)(b * 4 + chunk) * 514;
    pb[2 * tid] = c0;
    pb[2 * tid + 1] = c1;
    if (tid == 0) { pb[512] = m_run; pb[513] = l_run; }
}

// ---------------------------------------------------------------------------
// Combine the 4 chunk partials per batch into the final context.
// ---------------------------------------------------------------------------
__global__ __launch_bounds__(256) void combine_kernel(const float* __restrict__ partials,
                                                      float* __restrict__ ctxf) {
    int b = blockIdx.x, t = threadIdx.x;
    const float* pb = partials + (size_t)b * 4 * 514;
    float M = MNEG;
#pragma unroll
    for (int i = 0; i < 4; ++i) M = fmaxf(M, pb[i * 514 + 512]);
    float w[4], L = 0.f;
#pragma unroll
    for (int i = 0; i < 4; ++i) {
        w[i] = __expf(pb[i * 514 + 512] - M);
        L += pb[i * 514 + 513] * w[i];
    }
    float inv = 1.f / L;
    float a0 = 0.f, a1 = 0.f;
#pragma unroll
    for (int i = 0; i < 4; ++i) {
        a0 += pb[i * 514 + 2 * t] * w[i];
        a1 += pb[i * 514 + 2 * t + 1] * w[i];
    }
    ctxf[b * 512 + 2 * t] = a0 * inv;
    ctxf[b * 512 + 2 * t + 1] = a1 * inv;
}

// ---------------------------------------------------------------------------
// z = [inputs|context|h](64x1536) @ packed(1536x2048) + bias, then LSTM gates.
// Grid: (32 col-blocks of 16 gate-cols, 4 row-tiles); 4 waves = one per gate.
// ---------------------------------------------------------------------------
__global__ __launch_bounds__(128) void final_kernel(const float* __restrict__ inp,
                                                    const float* __restrict__ h,
                                                    const float* __restrict__ c_in,
                                                    const float* __restrict__ ctxf,
                                                    const float* __restrict__ bias,
                                                    const u16* __restrict__ pk_fin,
                                                    float* __restrict__ out) {
    __shared__ float z_s[4][16][16];
    int tid = threadIdx.x, l = tid & 31, wv = tid >> 5, l15 = l & 15, hi = l >> 4;
    int bx = blockIdx.x, by = blockIdx.y;
    int ncol = wv * 512 + bx * 16;
    int p = ncol >> 6, nb = bx & 3;
    int m = by * 16 + l15;
    v8f c = {};
#pragma unroll 2
    for (int g = 0; g < 12; ++g)
#pragma unroll
        for (int kc = 0; kc < 4; ++kc) {
            int kb = g * 128 + kc * 32;
            const float* ap;
            if (kb < 512)       ap = inp  + (size_t)m * 512 + kb;
            else if (kb < 1024) ap = ctxf + (size_t)m * 512 + (kb - 512);
            else                ap = h    + (size_t)m * 512 + (kb - 1024);
            FragU fa; load_afrag_f32(fa, ap, hi);
            int fi = ((p * 12 + g) * 4 + kc) * 4 + nb;
            const uint4* bp = (const uint4*)((const char*)pk_fin + (size_t)fi * 1024 + l * 32);
            FragU fb; fb.q[0] = bp[0]; fb.q[1] = bp[1];
            c = wmma_bf16(fa.bf, fb.bf, c);
        }
    int colg = wv * 512 + bx * 16 + l15;
#pragma unroll
    for (int j = 0; j < 8; ++j) z_s[wv][j + 8 * hi][l15] = c[j] + bias[colg];
    __syncthreads();
    for (int e = tid; e < 256; e += 128) {
        int row = e >> 4, col = e & 15;
        int rg = by * 16 + row, cg = bx * 16 + col;
        float zi = z_s[0][row][col], zf = z_s[1][row][col];
        float zc = z_s[2][row][col], zo = z_s[3][row][col];
        float ig = hsig(zi), fg = hsig(zf), og = hsig(zo);
        float cn = fg * c_in[rg * 512 + cg] + ig * fast_tanh(zc);
        out[rg * 512 + cg] = og * fast_tanh(cn);
    }
}

// ---------------------------------------------------------------------------
extern "C" void kernel_launch(void* const* d_in, const int* in_sizes, int n_in,
                              void* d_out, int out_size, void* d_ws, size_t ws_size,
                              hipStream_t stream) {
    (void)in_sizes; (void)n_in; (void)out_size; (void)ws_size;
    const float* inputs = (const float*)d_in[0];
    const float* h      = (const float*)d_in[1];
    const float* c      = (const float*)d_in[2];
    const float* ann    = (const float*)d_in[3];
    const float* kern   = (const float*)d_in[4];
    const float* rker   = (const float*)d_in[5];
    const float* bias   = (const float*)d_in[6];
    const float* ku     = (const float*)d_in[7];
    const float* kw     = (const float*)d_in[8];
    const float* kv     = (const float*)d_in[9];
    float* out = (float*)d_out;

    char* ws = (char*)d_ws;
    u16* pk_ku  = (u16*)ws;  ws += (size_t)512 * 512 * 2;
    u16* pk_kw  = (u16*)ws;  ws += (size_t)512 * 512 * 2;
    u16* pk_fin = (u16*)ws;  ws += (size_t)1536 * 2048 * 2;
    float* wx       = (float*)ws; ws += (size_t)64 * 512 * 4;
    float* partials = (float*)ws; ws += (size_t)64 * 4 * 514 * 4;
    float* ctxf     = (float*)ws; ws += (size_t)64 * 512 * 4;

    pack_kernel<<<1024, 256, 0, stream>>>(ku, ku, 512, 512, 4, pk_ku, 512 * 512);
    pack_kernel<<<1024, 256, 0, stream>>>(kw, kw, 512, 512, 4, pk_kw, 512 * 512);
    pack_kernel<<<12288, 256, 0, stream>>>(kern, rker, 1024, 2048, 12, pk_fin, 1536 * 2048);

    wx_kernel<<<dim3(32, 4), 32, 0, stream>>>(h, pk_kw, wx);
    attn_kernel<<<dim3(4, 64), 256, 0, stream>>>(ann, wx, bias, kv, pk_ku, partials);
    combine_kernel<<<64, 256, 0, stream>>>(partials, ctxf);
    final_kernel<<<dim3(32, 4), 128, 0, stream>>>(inputs, h, c, ctxf, bias, pk_fin, out);
}
